// ConvBN2d_if_83322365542499
// MI455X (gfx1250) — matrix operand order repaired and checked by reading, not compiled, not run on hardware
//
#include <hip/hip_runtime.h>
#include <math.h>

// ---------------- problem constants ----------------
#define NB   32      // batch
#define TT   10      // timesteps
#define CC   128     // channels (in == out)
#define HH   32
#define WW   32
#define PIX  (HH*WW)            // 1024
#define PH   34                 // padded spatial extent (1-px zero halo)
#define PIMG ((size_t)PH*PH*CC) // halves per padded (n,t) image = 147968
#define THETA 1.0f

// ---------------- vector types ----------------
typedef __attribute__((ext_vector_type(16))) _Float16 v16h;
typedef __attribute__((ext_vector_type(8)))  _Float16 v8h;
typedef __attribute__((ext_vector_type(8)))  float    v8f;
typedef __attribute__((ext_vector_type(4)))  float    v4f;

// =====================================================================
// Kernel 1: folded bias  b_fold = (conv_b - mean) * gamma/sqrt(var) + beta
// =====================================================================
__global__ void bfold_kernel(const float* __restrict__ conv_b,
                             const float* __restrict__ gamma,
                             const float* __restrict__ beta,
                             const float* __restrict__ mean,
                             const float* __restrict__ var,
                             float* __restrict__ bfold) {
    int c = threadIdx.x;
    if (c < CC) {
        float ratio = gamma[c] / sqrtf(var[c]);      // no eps (matches ref fold)
        bfold[c] = (conv_b[c] - mean[c]) * ratio + beta[c];
    }
}

// =====================================================================
// Kernel 2: pack folded weights into WMMA B-fragment layout (f16)
//   Wp[kk][s][ct][lane][j] ; lane L: N = ct*16+(L&15), half j:
//   K(cin) = s*32 + (j>>3)*16 + (L>>4)*8 + (j&7)   (ISA B 32x16 16-bit layout)
// =====================================================================
#define WPACK_ELEMS (9*4*8*32*16)   // 147456
__global__ void wpack_kernel(const float* __restrict__ conv_w,   // [COUT][CIN][3][3]
                             const float* __restrict__ gamma,
                             const float* __restrict__ var,
                             _Float16* __restrict__ Wp) {
    int e = blockIdx.x * 256 + threadIdx.x;
    if (e >= WPACK_ELEMS) return;
    int j  =  e        & 15;
    int L  = (e >> 4)  & 31;
    int ct = (e >> 9)  & 7;
    int s  = (e >> 12) & 3;
    int kk =  e >> 14;           // 0..8
    int ky = kk / 3, kx = kk % 3;
    int cin  = s * 32 + (j >> 3) * 16 + (L >> 4) * 8 + (j & 7);
    int cout = ct * 16 + (L & 15);
    float ratio = gamma[cout] / sqrtf(var[cout]);
    float wv = conv_w[((cout * CC + cin) * 3 + ky) * 3 + kx] * ratio;
    Wp[e] = (_Float16)wv;
}

// =====================================================================
// Kernel 3: zero the 1-px halo of the padded f16 image buffer.
//   grid = (NT, 132 border cells), 64 threads = 64 dwords = 128 halves.
// =====================================================================
__global__ __launch_bounds__(64) void halo_zero_kernel(_Float16* __restrict__ Xp) {
    int nt   = blockIdx.x;
    int cell = blockIdx.y;       // 0..131
    int y, x;
    if      (cell < 34)  { y = 0;              x = cell;       }
    else if (cell < 68)  { y = PH - 1;         x = cell - 34;  }
    else if (cell < 100) { y = cell - 68 + 1;  x = 0;          }
    else                 { y = cell - 100 + 1; x = PH - 1;     }
    unsigned int* p = (unsigned int*)(Xp + (size_t)nt * PIMG + (size_t)(y * PH + x) * CC);
    p[threadIdx.x] = 0u;
}

// =====================================================================
// Kernel 4: NCHW f32 spikes -> padded channels-last f16
//   Xp[(n*T+t)][1+h][1+w][cin].  Tiled LDS transpose: coalesced f32
//   reads along p, coalesced f16 writes along cin.
// =====================================================================
__global__ __launch_bounds__(256) void xform_kernel(const float* __restrict__ in,
                                                    _Float16* __restrict__ out) {
    __shared__ float tile[32][33];
    int b     = blockIdx.x;
    int ctile =  b       & 3;
    int ptile = (b >> 2) & 31;
    int nt    =  b >> 7;
    int tid   = threadIdx.x;

    int pl  = tid & 31;
    int cl0 = tid >> 5;                       // 0..7
    size_t inBase = ((size_t)nt * CC + (size_t)ctile * 32) * PIX + (size_t)ptile * 32;
#pragma unroll
    for (int i = 0; i < 4; ++i) {
        int cl = cl0 + i * 8;
        tile[cl][pl] = in[inBase + (size_t)cl * PIX + pl];
    }
    __syncthreads();

    int cl  = tid & 31;
    int pl0 = tid >> 5;
#pragma unroll
    for (int i = 0; i < 4; ++i) {
        int plc = pl0 + i * 8;
        int p = ptile * 32 + plc;
        int y = p >> 5, x = p & 31;
        out[(size_t)nt * PIMG + (size_t)((y + 1) * PH + (x + 1)) * CC
            + ctile * 32 + cl] = (_Float16)tile[cl][plc];
    }
}

// =====================================================================
// Kernel 5: implicit-GEMM 3x3 conv via WMMA + in-register IF scan.
//   Halo-padded input -> no branches: body is global_load_b128 with
//   immediate offsets + v_wmma.  Depth-2 software pipeline with
//   sched_barrier(0) pins issue order (loads for q+2 issued before
//   WMMA q) so the waitcnt pass can use partial loadcnt waits: every
//   WMMA runs with 4 A-loads in flight.  One wave owns (n, 16-pixel
//   tile, 16-cout tile); 10 accumulator fragments carry all timesteps
//   and the IF recurrence runs entirely in registers.
// =====================================================================
__global__ __launch_bounds__(256) void snn_conv_if_kernel(
    const _Float16* __restrict__ X,     // padded [N*T][34][34][CC] f16
    const _Float16* __restrict__ Wp,    // packed B fragments
    const float*    __restrict__ bfold, // [CC]
    float* __restrict__ out_st,         // [N][T][CC][PIX]
    float* __restrict__ out_sc)         // [N][CC][PIX]
{
    const int lane = threadIdx.x & 31;
    const int wid  = blockIdx.x * 8 + (threadIdx.x >> 5);
    const int ct   =  wid       & 7;    // cout tile (fastest: waves share A)
    const int st   = (wid >> 3) & 63;   // spatial tile
    const int n    =  wid >> 9;

    const int m    = lane & 15;         // A-matrix row / spatial idx in tile
    const int hi   = lane >> 4;
    const int y    = st >> 1;
    const int x0   = (st & 1) << 4;
    const int cout = ct * 16 + m;

    v8f acc[TT] = {};                   // pots accumulators, one per timestep

    // Base pointer: tap (ky=0,kx=0) == padded coords (y, x0+m); hi-dependent
    // k-offset folded in so every load is base + compile-time immediate.
    const _Float16* P0 = X + (size_t)n * TT * PIMG
                           + (size_t)(y * PH + x0 + m) * CC + hi * 8;
    const _Float16* wbase = Wp + (size_t)ct * 512 + (size_t)lane * 16;

    for (int ky = 0; ky < 3; ++ky) {
        for (int kx = 0; kx < 3; ++kx) {
            const int kk = ky * 3 + kx;
            const _Float16* pt = P0 + (size_t)(ky * PH + kx) * CC;

            // All 4 B fragments for this tap (each reused by 10 WMMAs)
            v16h bf[4];
#pragma unroll
            for (int s = 0; s < 4; ++s)
                bf[s] = *(const v16h*)(wbase + (size_t)(kk * 4 + s) * 4096);

            // Depth-2 software pipeline over flat q = s*10 + t.
            // Under full unroll A0/A1 are SSA values: ~3 buffers live.
            v8h A0[40], A1[40];
            {
                A0[0] = *(const v8h*)(pt);
                A1[0] = *(const v8h*)(pt + 16);
                const _Float16* a1p = pt + PIMG;          // q=1: s=0,t=1
                A0[1] = *(const v8h*)(a1p);
                A1[1] = *(const v8h*)(a1p + 16);
            }
#pragma unroll
            for (int q = 0; q < 40; ++q) {
                const int s = q / 10, t = q % 10;
                if (q + 2 < 40) {
                    const int s2 = (q + 2) / 10, t2 = (q + 2) % 10;
                    const _Float16* ar = pt + (size_t)t2 * PIMG + s2 * 32;
                    A0[q + 2] = *(const v8h*)(ar);        // K = s*32+hi*8    ..+8
                    A1[q + 2] = *(const v8h*)(ar + 16);   // K = s*32+16+hi*8 ..+8
                }
                // Pin issue order: the q+2 loads must precede this WMMA so
                // only a partial s_wait_loadcnt is needed (4 in flight).
                __builtin_amdgcn_sched_barrier(0);
                v16h afrag = __builtin_shufflevector(A0[q], A1[q],
                    0,1,2,3,4,5,6,7, 8,9,10,11,12,13,14,15);
                acc[t] = __builtin_amdgcn_wmma_f32_16x16x32_f16(
                    false, afrag, false, bf[s], (short)0, acc[t], false, false);
            }
        }
    }

    // ---- IF dynamics in registers: mem0 = b_fold, spike = (mem > theta),
    // ---- reset by subtraction, count spikes over T.
    const float b0 = bfold[cout];
    float mem[8], cnt[8];
#pragma unroll
    for (int r = 0; r < 8; ++r) { mem[r] = b0; cnt[r] = 0.f; }

    const int pbase = y * WW + x0 + hi * 8;     // 8 contiguous pixels per lane
    float* stBase = out_st + ((size_t)n * TT * CC + cout) * PIX + pbase;
#pragma unroll
    for (int t = 0; t < TT; ++t) {
        float sp[8];
#pragma unroll
        for (int r = 0; r < 8; ++r) {
            mem[r] += acc[t][r];
            float s_ = (mem[r] > THETA) ? 1.0f : 0.0f;
            mem[r] -= s_ * THETA;
            cnt[r] += s_;
            sp[r] = s_;
        }
        float* o = stBase + (size_t)t * CC * PIX;
        *(v4f*)(o)     = (v4f){sp[0], sp[1], sp[2], sp[3]};
        *(v4f*)(o + 4) = (v4f){sp[4], sp[5], sp[6], sp[7]};
    }
    // forward value of output_features_sc == spike_count (straight-through)
    float* oc = out_sc + ((size_t)n * CC + cout) * PIX + pbase;
    *(v4f*)(oc)     = (v4f){cnt[0], cnt[1], cnt[2], cnt[3]};
    *(v4f*)(oc + 4) = (v4f){cnt[4], cnt[5], cnt[6], cnt[7]};
}

// =====================================================================
// launch
// =====================================================================
extern "C" void kernel_launch(void* const* d_in, const int* in_sizes, int n_in,
                              void* d_out, int out_size, void* d_ws, size_t ws_size,
                              hipStream_t stream) {
    const float* in_st  = (const float*)d_in[0];   // [N,T,CIN,H,W] spikes
    // d_in[1] (input_features_sc) unused: forward sc output == spike count
    const float* conv_w = (const float*)d_in[2];
    const float* conv_b = (const float*)d_in[3];
    const float* gamma  = (const float*)d_in[4];
    const float* beta   = (const float*)d_in[5];
    const float* mean   = (const float*)d_in[6];
    const float* var    = (const float*)d_in[7];

    // workspace layout (all 256B-aligned)
    char* ws = (char*)d_ws;
    _Float16* Wp    = (_Float16*)(ws);                 // 294,912 B
    float*    bfold = (float*)(ws + 294912);           //     512 B
    _Float16* Xp    = (_Float16*)(ws + 295424);        // 94,699,520 B (padded)

    float* out_st = (float*)d_out;                                   // 41,943,040 f32
    float* out_sc = out_st + (size_t)NB * TT * CC * PIX;             //  4,194,304 f32

    hipLaunchKernelGGL(bfold_kernel, dim3(1), dim3(128), 0, stream,
                       conv_b, gamma, beta, mean, var, bfold);
    hipLaunchKernelGGL(wpack_kernel, dim3((WPACK_ELEMS + 255) / 256), dim3(256), 0, stream,
                       conv_w, gamma, var, Wp);
    hipLaunchKernelGGL(halo_zero_kernel, dim3(NB * TT, 132), dim3(64), 0, stream, Xp);
    hipLaunchKernelGGL(xform_kernel, dim3(NB * TT * 32 * 4), dim3(256), 0, stream,
                       in_st, Xp);
    // waves: N(32) * spatial tiles(64) * cout tiles(8) = 16384 -> 2048 blocks x 8 waves
    hipLaunchKernelGGL(snn_conv_if_kernel, dim3(2048), dim3(256), 0, stream,
                       Xp, Wp, bfold, out_st, out_sc);
}
